// BandsVQAutoencoder_81750407512159
// MI455X (gfx1250) — matrix operand order repaired and checked by reading, not compiled, not run on hardware
//
#include <hip/hip_runtime.h>

// ---------------------------------------------------------------------------
// BandsVQAutoencoder for MI455X (gfx1250, wave32, WMMA 16x16x32 f16 -> f32 acc)
//
// d_out layout (floats, reference return order, flattened):
//   bands_hat : [0        , 1024000)
//   z_e       : [1024000  , 5120000)
//   z_q       : [5120000  , 9216000)
//   idx(float): [9216000  , 9344000)
//   vq_loss   : [9344000]
// ---------------------------------------------------------------------------

typedef __attribute__((ext_vector_type(16))) _Float16 v16h;
typedef __attribute__((ext_vector_type(8)))  float    v8f;

#define WMMA_F16(a, b, c) \
  __builtin_amdgcn_wmma_f32_16x16x32_f16(false, (a), false, (b), (short)0, (c), false, false)

#define NTOK   16000
#define LATENT 256
#define HIDDEN 512
#define NBANDS 64
#define G      8
#define KCODE  1024
#define GD     32
#define NPART  4000   // gather partial-sum blocks

// CDNA5 ISA 7.12.2: 16-bit A-matrix 16x32 mapping (branchless).
// Row m lives in lanes m and m+16; lanes 0..15 hold K in {0..7,16..23},
// lanes 16..31 hold {8..15,24..31}.
//   hi = (kl>>3)&1            (which 16-lane group)
//   e  = (kl&7) | ((kl>>4)<<3) (element index within the 16-half fragment)
__device__ __forceinline__ int amap_hi(int kl) { return (kl >> 3) & 1; }
__device__ __forceinline__ int amap_e(int kl)  { return (kl & 7) | ((kl >> 4) << 3); }

// ---------------------------------------------------------------------------
// Weight repack: W (K x N, f32 row-major) -> f16 B-fragments.
// B-matrix (16-bit, 32x16): lane n = lane%16; lanes 0..15 hold K in {0..7,16..23},
// lanes 16..31 hold {8..15,24..31} (same element order as the A mapping).
// Fragment layout: [kt][nt][lane][e], 16 halfs contiguous per lane.
// ---------------------------------------------------------------------------
__global__ __launch_bounds__(256) void pack_w(const float* __restrict__ W,
                                              _Float16* __restrict__ out,
                                              int N, int NT, int total) {
  int t = blockIdx.x * 256 + threadIdx.x;
  if (t >= total) return;
  int e    = t & 15;
  int lane = (t >> 4) & 31;
  int q    = t >> 9;
  int nt   = q % NT;
  int kt   = q / NT;
  int k = kt * 32 + ((lane >> 4) << 3) + (e & 7) + ((e >> 3) << 4);
  int n = nt * 16 + (lane & 15);
  out[t] = (_Float16)W[k * N + n];
}

// Codebooks (G,K,GD) -> f16 B-fragments, layout [g][tile][lane][e]
__global__ __launch_bounds__(256) void pack_cb(const float* __restrict__ cb,
                                               _Float16* __restrict__ out) {
  int t = blockIdx.x * 256 + threadIdx.x;        // 262144 total
  int e    = t & 15;
  int lane = (t >> 4) & 31;
  int tt   = (t >> 9) & 63;
  int g    = t >> 15;
  int code = tt * 16 + (lane & 15);
  int k    = ((lane >> 4) << 3) + (e & 7) + ((e >> 3) << 4);
  out[t] = (_Float16)cb[((size_t)(g * KCODE + code)) * GD + k];
}

// cn[g][k] = sum_d cb^2 (exact f32, keeps argmin robust vs f16 cross term)
__global__ __launch_bounds__(256) void pack_cn(const float* __restrict__ cb,
                                               float* __restrict__ cn) {
  int t = blockIdx.x * 256 + threadIdx.x;        // 8192 total
  const float* row = cb + (size_t)t * GD;
  float s = 0.f;
  #pragma unroll
  for (int i = 0; i < GD; ++i) { float v = row[i]; s += v * v; }
  cn[t] = s;
}

// ---------------------------------------------------------------------------
// Encoder: z_e = relu(x@W1+b1)@W2+b2. 32-token tile / block, 8 waves.
// ---------------------------------------------------------------------------
__global__ __launch_bounds__(256) void enc_kernel(
    const float* __restrict__ bands, const float* __restrict__ b1,
    const float* __restrict__ b2, const _Float16* __restrict__ w1f,
    const _Float16* __restrict__ w2f, float* __restrict__ zep) {
  __shared__ alignas(32) _Float16 xf[2][2][32][16];   // x A-frags  (4 KB)
  __shared__ alignas(32) _Float16 hf[2][16][32][16];  // h A-frags (32 KB)
  const int tid = threadIdx.x, wid = tid >> 5, lane = tid & 31;
  const int ln = lane & 15, hiL = lane >> 4;
  const int tok0 = blockIdx.x * 32;

  // stage bands tile straight into A-fragment order (f16)
  #pragma unroll
  for (int i = 0; i < 8; ++i) {
    int idx = tid + 256 * i;                    // 32x64 = 2048 elems
    int ml = idx >> 6, k = idx & 63;
    int kl = k & 31;
    xf[ml >> 4][k >> 5][(ml & 15) + 16 * amap_hi(kl)][amap_e(kl)] =
        (_Float16)bands[(size_t)(tok0 + ml) * NBANDS + k];
  }
  __syncthreads();

  // layer1: 64 (msub,nt) jobs, K=64 (2 wmma each)
  for (int i = 0; i < 8; ++i) {
    int job = wid + 8 * i;
    int msub = job >> 5, nt = job & 31;
    float bv = b1[nt * 16 + ln];
    v8f acc;
    #pragma unroll
    for (int r = 0; r < 8; ++r) acc[r] = bv;
    #pragma unroll
    for (int kt = 0; kt < 2; ++kt) {
      v16h a = *(const v16h*)&xf[msub][kt][lane][0];
      v16h b = *(const v16h*)(w1f + (size_t)(((kt * 32 + nt) * 32) + lane) * 16);
      acc = WMMA_F16(a, b, acc);
    }
    // store to hf in A-frag order; mapping is branchless + strength-reduced
    int kl = ((nt & 1) << 4) + ln;              // = (nt*16+ln) & 31
    _Float16* hp = &hf[msub][nt >> 1][16 * amap_hi(kl) + 8 * hiL][amap_e(kl)];
    #pragma unroll
    for (int r = 0; r < 8; ++r) {
      float v = fmaxf(acc[r], 0.f);             // ReLU
      hp[r * 16] = (_Float16)v;                 // +1 row = +16 halfs
    }
  }
  __syncthreads();

  // layer2: 32 jobs, K=512 (16 wmma each); write z_e (f32)
  for (int i = 0; i < 4; ++i) {
    int job = wid + 8 * i;
    int msub = job >> 4, nt = job & 15;
    float bv = b2[nt * 16 + ln];
    v8f acc;
    #pragma unroll
    for (int r = 0; r < 8; ++r) acc[r] = bv;
    for (int kt = 0; kt < 16; ++kt) {
      v16h a = *(const v16h*)&hf[msub][kt][lane][0];
      v16h b = *(const v16h*)(w2f + (size_t)(((kt * 16 + nt) * 32) + lane) * 16);
      acc = WMMA_F16(a, b, acc);
    }
    float* zp = zep + (size_t)(tok0 + msub * 16 + 8 * hiL) * LATENT + nt * 16 + ln;
    #pragma unroll
    for (int r = 0; r < 8; ++r) zp[(size_t)r * LATENT] = acc[r];
  }
}

// ---------------------------------------------------------------------------
// VQ argmin: 16-token tile / block; wave w handles group w. One WMMA per
// 16-code tile (K=GD=32). Score = cn - 2*cross (zn constant per row).
// ---------------------------------------------------------------------------
__global__ __launch_bounds__(256) void vq_kernel(
    const float* __restrict__ zep, const _Float16* __restrict__ cbf,
    const float* __restrict__ cn, float* __restrict__ idxf,
    int* __restrict__ idxi) {
  __shared__ alignas(32) _Float16 zf[G][32][16];   // z_e A-frags per group
  const int tid = threadIdx.x, g = tid >> 5, lane = tid & 31;
  const int ln = lane & 15, hiL = lane >> 4;
  const int tok0 = blockIdx.x * 16;

  #pragma unroll
  for (int i = 0; i < 16; ++i) {
    int idx = tid + 256 * i;                     // 16x256 = 4096 elems
    int m = idx >> 8, c = idx & 255;
    int kl = c & 31;
    zf[c >> 5][m + 16 * amap_hi(kl)][amap_e(kl)] =
        (_Float16)zep[(size_t)(tok0 + m) * LATENT + c];
  }
  __syncthreads();

  v16h a = *(const v16h*)&zf[g][lane][0];
  float bestv[8]; int besti[8];
  #pragma unroll
  for (int r = 0; r < 8; ++r) { bestv[r] = 3.0e38f; besti[r] = 0; }

  const _Float16* cbg = cbf + (size_t)g * 64 * 32 * 16;
  const float* cng = cn + g * KCODE;
  for (int t = 0; t < 64; ++t) {
    v16h b = *(const v16h*)(cbg + (size_t)(t * 32 + lane) * 16);
    v8f acc = {};
    acc = WMMA_F16(a, b, acc);
    int code = t * 16 + ln;
    float cv = cng[code];
    #pragma unroll
    for (int r = 0; r < 8; ++r) {
      float s = cv - 2.0f * acc[r];
      if (s < bestv[r]) { bestv[r] = s; besti[r] = code; }
    }
  }
  // butterfly argmin across the 16 lanes holding each row (masks<16 keep the
  // two half-wave row groups separate); tie-break: smaller index (argmin-first)
  #pragma unroll
  for (int m = 1; m <= 8; m <<= 1) {
    #pragma unroll
    for (int r = 0; r < 8; ++r) {
      float ov = __shfl_xor(bestv[r], m, 32);
      int   oi = __shfl_xor(besti[r], m, 32);
      if (ov < bestv[r] || (ov == bestv[r] && oi < besti[r])) {
        bestv[r] = ov; besti[r] = oi;
      }
    }
  }
  if (ln == 0) {
    #pragma unroll
    for (int r = 0; r < 8; ++r) {
      int row = tok0 + r + 8 * hiL;
      idxf[row * G + g] = (float)besti[r];
      idxi[row * G + g] = besti[r];
    }
  }
}

// ---------------------------------------------------------------------------
// Gather z_q = cb[idx], accumulate per-block sq-diff partials (deterministic).
// One thread per (token, group, 4-float chunk): 16000*8*8 threads.
// ---------------------------------------------------------------------------
__global__ __launch_bounds__(256) void gather_kernel(
    const float* __restrict__ zep, const float* __restrict__ cb,
    const int* __restrict__ idxi, float* __restrict__ zqp,
    float* __restrict__ partials) {
  __shared__ float red[256];
  int t  = blockIdx.x * 256 + threadIdx.x;
  int n  = t >> 6;
  int rem = t & 63;
  int g  = rem >> 3, dq = rem & 7;
  int id = idxi[n * G + g];
  const float4 c4 = ((const float4*)(cb + ((size_t)(g * KCODE + id)) * GD))[dq];
  const float4 z4 = ((const float4*)(zep + (size_t)n * LATENT + g * GD))[dq];
  ((float4*)(zqp + (size_t)n * LATENT + g * GD))[dq] = c4;
  float dx = c4.x - z4.x, dy = c4.y - z4.y, dz = c4.z - z4.z, dw = c4.w - z4.w;
  red[threadIdx.x] = dx * dx + dy * dy + dz * dz + dw * dw;
  __syncthreads();
  for (int s = 128; s > 0; s >>= 1) {
    if (threadIdx.x < s) red[threadIdx.x] += red[threadIdx.x + s];
    __syncthreads();
  }
  if (threadIdx.x == 0) partials[blockIdx.x] = red[0];
}

// vq_loss = 2*BETA * sum_sq / (N*GD) = 0.5 * total / 512000
__global__ __launch_bounds__(256) void loss_kernel(
    const float* __restrict__ partials, float* __restrict__ loss, int np) {
  __shared__ float red[256];
  float s = 0.f;
  for (int i = threadIdx.x; i < np; i += 256) s += partials[i];
  red[threadIdx.x] = s;
  __syncthreads();
  for (int st = 128; st > 0; st >>= 1) {
    if (threadIdx.x < st) red[threadIdx.x] += red[threadIdx.x + st];
    __syncthreads();
  }
  if (threadIdx.x == 0) *loss = red[0] * (0.5f / 512000.0f);
}

// ---------------------------------------------------------------------------
// Decoder: bands_hat = relu(z_q@W1+b1)@W2+b2. 32-token tile / block.
// ---------------------------------------------------------------------------
__global__ __launch_bounds__(256) void dec_kernel(
    const float* __restrict__ zqp, const float* __restrict__ b1,
    const float* __restrict__ b2, const _Float16* __restrict__ w1f,
    const _Float16* __restrict__ w2f, float* __restrict__ bh) {
  __shared__ alignas(32) _Float16 xf[2][8][32][16];   // z_q A-frags (16 KB)
  __shared__ alignas(32) _Float16 hf[2][16][32][16];  // h   A-frags (32 KB)
  const int tid = threadIdx.x, wid = tid >> 5, lane = tid & 31;
  const int ln = lane & 15, hiL = lane >> 4;
  const int tok0 = blockIdx.x * 32;

  #pragma unroll
  for (int i = 0; i < 32; ++i) {
    int idx = tid + 256 * i;                    // 32x256 = 8192 elems
    int ml = idx >> 8, k = idx & 255;
    int kl = k & 31;
    xf[ml >> 4][k >> 5][(ml & 15) + 16 * amap_hi(kl)][amap_e(kl)] =
        (_Float16)zqp[(size_t)(tok0 + ml) * LATENT + k];
  }
  __syncthreads();

  // layer1: 64 jobs, K=256 (8 wmma each)
  for (int i = 0; i < 8; ++i) {
    int job = wid + 8 * i;
    int msub = job >> 5, nt = job & 31;
    float bv = b1[nt * 16 + ln];
    v8f acc;
    #pragma unroll
    for (int r = 0; r < 8; ++r) acc[r] = bv;
    #pragma unroll
    for (int kt = 0; kt < 8; ++kt) {
      v16h a = *(const v16h*)&xf[msub][kt][lane][0];
      v16h b = *(const v16h*)(w1f + (size_t)(((kt * 32 + nt) * 32) + lane) * 16);
      acc = WMMA_F16(a, b, acc);
    }
    int kl = ((nt & 1) << 4) + ln;
    _Float16* hp = &hf[msub][nt >> 1][16 * amap_hi(kl) + 8 * hiL][amap_e(kl)];
    #pragma unroll
    for (int r = 0; r < 8; ++r) {
      float v = fmaxf(acc[r], 0.f);
      hp[r * 16] = (_Float16)v;
    }
  }
  __syncthreads();

  // layer2: 8 jobs (1 per wave), K=512 (16 wmma)
  {
    int job = wid;
    int msub = job >> 2, nt = job & 3;
    float bv = b2[nt * 16 + ln];
    v8f acc;
    #pragma unroll
    for (int r = 0; r < 8; ++r) acc[r] = bv;
    for (int kt = 0; kt < 16; ++kt) {
      v16h a = *(const v16h*)&hf[msub][kt][lane][0];
      v16h b = *(const v16h*)(w2f + (size_t)(((kt * 4 + nt) * 32) + lane) * 16);
      acc = WMMA_F16(a, b, acc);
    }
    float* bp = bh + (size_t)(tok0 + msub * 16 + 8 * hiL) * NBANDS + nt * 16 + ln;
    #pragma unroll
    for (int r = 0; r < 8; ++r) bp[(size_t)r * NBANDS] = acc[r];
  }
}

// ---------------------------------------------------------------------------
extern "C" void kernel_launch(void* const* d_in, const int* in_sizes, int n_in,
                              void* d_out, int out_size, void* d_ws, size_t ws_size,
                              hipStream_t stream) {
  (void)in_sizes; (void)n_in; (void)out_size; (void)ws_size;
  const float* bands  = (const float*)d_in[0];
  const float* enc_w1 = (const float*)d_in[1];
  const float* enc_b1 = (const float*)d_in[2];
  const float* enc_w2 = (const float*)d_in[3];
  const float* enc_b2 = (const float*)d_in[4];
  const float* cb     = (const float*)d_in[5];
  const float* dec_w1 = (const float*)d_in[6];
  const float* dec_b1 = (const float*)d_in[7];
  const float* dec_w2 = (const float*)d_in[8];
  const float* dec_b2 = (const float*)d_in[9];

  float* out   = (float*)d_out;
  float* bh    = out;
  float* zep   = out + 1024000;
  float* zqp   = out + 5120000;
  float* idxf  = out + 9216000;
  float* lossp = out + 9344000;

  // workspace layout (~1.74 MB), all offsets 256B-aligned
  char* ws = (char*)d_ws;
  _Float16* w1f = (_Float16*)(ws + 0);         //  32768 halfs ( 64 KB)
  _Float16* w2f = (_Float16*)(ws + 65536);     // 131072 halfs (256 KB)
  _Float16* d1f = (_Float16*)(ws + 327680);    // 131072 halfs (256 KB)
  _Float16* d2f = (_Float16*)(ws + 589824);    //  32768 halfs ( 64 KB)
  _Float16* cbf = (_Float16*)(ws + 655360);    // 262144 halfs (512 KB)
  float*    cn  = (float*)(ws + 1179648);      //   8192 floats
  int*      idxi= (int*)(ws + 1212416);        // 128000 ints
  float*    part= (float*)(ws + 1724416);      //   4000 floats

  pack_w<<<128, 256, 0, stream>>>(enc_w1, w1f, 512, 32, 32768);
  pack_w<<<512, 256, 0, stream>>>(enc_w2, w2f, 256, 16, 131072);
  pack_w<<<512, 256, 0, stream>>>(dec_w1, d1f, 512, 32, 131072);
  pack_w<<<128, 256, 0, stream>>>(dec_w2, d2f, 64, 4, 32768);
  pack_cb<<<1024, 256, 0, stream>>>(cb, cbf);
  pack_cn<<<32, 256, 0, stream>>>(cb, cn);

  enc_kernel<<<NTOK / 32, 256, 0, stream>>>(bands, enc_b1, enc_b2, w1f, w2f, zep);
  vq_kernel<<<NTOK / 16, 256, 0, stream>>>(zep, cbf, cn, idxf, idxi);
  gather_kernel<<<NPART, 256, 0, stream>>>(zep, cb, idxi, zqp, part);
  loss_kernel<<<1, 256, 0, stream>>>(part, lossp, NPART);
  dec_kernel<<<NTOK / 32, 256, 0, stream>>>(zqp, dec_b1, dec_b2, d1f, d2f, bh);
}
